// Attention_cfged_88527865905334
// MI455X (gfx1250) — matrix operand order repaired and checked by reading, not compiled, not run on hardware
//
#include <hip/hip_runtime.h>

typedef _Float16 h16;
typedef __attribute__((ext_vector_type(16))) _Float16 v16h;
typedef __attribute__((ext_vector_type(8)))  _Float16 v8h;
typedef __attribute__((ext_vector_type(8)))  float    v8f;

#define WMMA_F16(a, b, c) __builtin_amdgcn_wmma_f32_16x16x32_f16( \
    false, (a), false, (b), (short)0, (c), false, false)

// ---------------- constants ----------------
// B=64, N=197, C=768, H=8, dq=64, dv=80, C_QK=1024, C_tot=1664
// M = B*N = 12608 = 788*16 = 394*32 exactly. Npad = 224 = 7*32.
#define SEQ      197
#define BATCH    64
#define HEADS    8
#define CIN      768
#define CTOT     1664
#define CQK      1024
#define DQ       64
#define DV       80
#define MROWS    12608
#define NPAD     224
#define SCALE    0.125f

// ---------------- WMMA fragment loaders ----------------
// A (16x32 f16, row-major source): lane row = rowBase+(lane&15); p = row ptr + kBase.
//   elems 0..7  -> K = 8*laneHi + e ; elems 8..15 -> K = 16 + 8*laneHi + e
__device__ __forceinline__ v16h load_frag_a(const h16* __restrict__ p, int laneHi, bool valid) {
    v8h a0 = *(const v8h*)(p + laneHi * 8);
    v8h a1 = *(const v8h*)(p + 16 + laneHi * 8);
    v16h r;
#pragma unroll
    for (int e = 0; e < 8; ++e) {
        r[e]     = valid ? a0[e] : (h16)0.0f;
        r[e + 8] = valid ? a1[e] : (h16)0.0f;
    }
    return r;
}

// B (32x16 f16): lane col = colBase+(lane&15); column's K contiguous in memory.
//   elems 0..7  -> K = 8*laneHi + e ; elems 8..15 -> K = 16 + 8*laneHi + e
__device__ __forceinline__ v16h load_frag_b(const h16* __restrict__ p, int laneHi, bool valid) {
    v8h b0 = *(const v8h*)(p + laneHi * 8);
    v8h b1 = *(const v8h*)(p + 16 + laneHi * 8);
    v16h r;
#pragma unroll
    for (int e = 0; e < 8; ++e) {
        r[e]     = valid ? b0[e] : (h16)0.0f;
        r[e + 8] = valid ? b1[e] : (h16)0.0f;
    }
    return r;
}

// Fragment set for a 32x64 wave tile at one K-step: 2 A-frags + 4 B-frags.
struct FragSet {
    v16h a[2];
    v16h b[4];
};

__device__ __forceinline__ void load_set(FragSet& s,
                                         const h16* __restrict__ arow0,
                                         const h16* __restrict__ arow1,
                                         const h16* __restrict__ bcol,
                                         int ldb, int kb, int laneHi) {
    s.a[0] = load_frag_a(arow0 + kb, laneHi, true);
    s.a[1] = load_frag_a(arow1 + kb, laneHi, true);
#pragma unroll
    for (int j = 0; j < 4; ++j)
        s.b[j] = load_frag_b(bcol + (size_t)j * 16 * ldb + kb, laneHi, true);
}

__device__ __forceinline__ void mma_set(v8f acc[8], const FragSet& s) {
#pragma unroll
    for (int mi = 0; mi < 2; ++mi)
#pragma unroll
        for (int j = 0; j < 4; ++j)
            acc[mi * 4 + j] = WMMA_F16(s.a[mi], s.b[j], acc[mi * 4 + j]);
}

// ---------------- conversion kernels ----------------
__global__ void cvt_x_kernel(const float* __restrict__ x, h16* __restrict__ xh) {
    size_t i = (size_t)blockIdx.x * blockDim.x + threadIdx.x;   // 9,682,944 exact
    xh[i] = (h16)x[i];
}

// W_qkv [768][1664] -> wt [1664][768]
__global__ void cvt_wqkv_kernel(const float* __restrict__ w, h16* __restrict__ wt) {
    size_t i = (size_t)blockIdx.x * blockDim.x + threadIdx.x;   // 1,277,952 exact
    int n = (int)(i / CIN);
    int k = (int)(i % CIN);
    wt[i] = (h16)w[(size_t)k * CTOT + n];
}

// W_proj [640][768] -> wt [768][640]
__global__ void cvt_wproj_kernel(const float* __restrict__ w, h16* __restrict__ wt) {
    size_t i = (size_t)blockIdx.x * blockDim.x + threadIdx.x;   // 491,520 exact
    int n = (int)(i / (HEADS * DV));
    int k = (int)(i % (HEADS * DV));
    wt[i] = (h16)w[(size_t)k * CIN + n];
}

// v_t[b][h][d][m] (m padded to 224 with zeros) from qkv_h
__global__ void build_vt_kernel(const h16* __restrict__ qkvh, h16* __restrict__ vt) {
    size_t i = (size_t)blockIdx.x * blockDim.x + threadIdx.x;   // 9,175,040 exact
    int m = (int)(i % NPAD);
    size_t t = i / NPAD;
    int d = (int)(t % DV);
    int bh = (int)(t / DV);
    int h = bh & 7, b = bh >> 3;
    h16 v = (h16)0.0f;
    if (m < SEQ) v = qkvh[(size_t)(b * SEQ + m) * CTOT + CQK + h * DV + d];
    vt[i] = v;
}

// ---------------- GEMM1: qkv = x @ W_qkv (f16 out) ----------------
// 32x64 per wave, 2-phase double-buffered K loop (K=768, 24 steps of 32).
// waves = 394 * 26 = 10244 -> blockDim 128, 2561 blocks exact.
__global__ __launch_bounds__(128) void gemm_qkv_kernel(
    const h16* __restrict__ xh, const h16* __restrict__ wt, h16* __restrict__ qkvh) {
    int wave = (int)((blockIdx.x * blockDim.x + threadIdx.x) >> 5);
    int lane = threadIdx.x & 31;
    int laneLo = lane & 15, laneHi = lane >> 4;
    int mpair = wave / 26;           // 0..393 (32 rows each)
    int ngrp  = wave % 26;           // 64 cols each

    const h16* arow0 = xh + (size_t)(mpair * 32 + laneLo) * CIN;
    const h16* arow1 = arow0 + (size_t)16 * CIN;
    const h16* bcol  = wt + (size_t)(ngrp * 64 + laneLo) * CIN;

    FragSet s0, s1;
    load_set(s0, arow0, arow1, bcol, CIN, 0, laneHi);
    load_set(s1, arow0, arow1, bcol, CIN, 32, laneHi);

    v8f acc[8] = {};
#pragma unroll 1
    for (int kb = 64; kb < CIN; kb += 64) {
        mma_set(acc, s0);
        load_set(s0, arow0, arow1, bcol, CIN, kb, laneHi);
        mma_set(acc, s1);
        load_set(s1, arow0, arow1, bcol, CIN, kb + 32, laneHi);
    }
    mma_set(acc, s0);
    mma_set(acc, s1);

#pragma unroll
    for (int mi = 0; mi < 2; ++mi)
#pragma unroll
        for (int j = 0; j < 4; ++j) {
            int col = ngrp * 64 + j * 16 + laneLo;
#pragma unroll
            for (int r = 0; r < 8; ++r) {
                int m = mpair * 32 + mi * 16 + 8 * laneHi + r;
                qkvh[(size_t)m * CTOT + col] = (h16)acc[mi * 4 + j][r];
            }
        }
}

// ---------------- fused attention: S = scale*qK^T, softmax, ctx = P@v ----------------
// 1 wave per block. grid = (13 row-tiles, H, B).
__global__ __launch_bounds__(32) void attn_fused_kernel(
    const h16* __restrict__ qkvh, const h16* __restrict__ vt,
    float* __restrict__ attn_out, h16* __restrict__ ctxh) {
    __shared__ float S[16][NPAD];

    int ntile = blockIdx.x;
    int h     = blockIdx.y;
    int b     = blockIdx.z;
    int lane  = threadIdx.x;
    int laneLo = lane & 15, laneHi = lane >> 4;

    // ---- scores: 13 m-tiles x 2 K-steps of WMMA ----
    int n = ntile * 16 + laneLo;
    bool nValid = n < SEQ;
    const h16* qrow = qkvh + (size_t)(b * SEQ + (nValid ? n : 0)) * CTOT + h * DQ;
    v16h qa0 = load_frag_a(qrow,      laneHi, nValid);
    v16h qa1 = load_frag_a(qrow + 32, laneHi, nValid);

    v8f acc[13];
#pragma unroll
    for (int mt = 0; mt < 13; ++mt) {
        int m = mt * 16 + laneLo;
        bool mValid = m < SEQ;
        const h16* krow = qkvh + (size_t)(b * SEQ + (mValid ? m : 0)) * CTOT + CQK / 2 + h * DQ;
        v16h b0 = load_frag_b(krow,      laneHi, mValid);
        v16h b1 = load_frag_b(krow + 32, laneHi, mValid);
        v8f c = {};
        c = WMMA_F16(qa0, b0, c);
        c = WMMA_F16(qa1, b1, c);
        acc[mt] = c;
    }
#pragma unroll
    for (int mt = 0; mt < 13; ++mt)
#pragma unroll
        for (int r = 0; r < 8; ++r)
            S[8 * laneHi + r][mt * 16 + laneLo] = acc[mt][r] * SCALE;
    __syncthreads();

    // ---- softmax: one thread per row ----
    if (lane < 16) {
        int rn = ntile * 16 + lane;
        if (rn < SEQ) {
            float mx = -3.0e38f;
            for (int m = 0; m < SEQ; ++m) mx = fmaxf(mx, S[lane][m]);
            float sum = 0.0f;
            for (int m = 0; m < SEQ; ++m) {
                float e = __expf(S[lane][m] - mx);
                S[lane][m] = e;
                sum += e;
            }
            float inv = 1.0f / sum;
            for (int m = 0; m < SEQ; ++m) S[lane][m] *= inv;
            for (int m = SEQ; m < NPAD; ++m) S[lane][m] = 0.0f;
        } else {
            for (int m = 0; m < NPAD; ++m) S[lane][m] = 0.0f;
        }
    }
    __syncthreads();

    // ---- write f32 attention probabilities (required output #2) ----
    {
        float* ap = attn_out + (size_t)(b * HEADS + h) * SEQ * SEQ;
        for (int idx = lane; idx < 16 * SEQ; idx += 32) {
            int r = idx / SEQ, m = idx - r * SEQ;
            int rn = ntile * 16 + r;
            if (rn < SEQ) ap[(size_t)rn * SEQ + m] = S[r][m];
        }
    }
    __syncthreads();

    // ---- ctx = P @ v : 7 K-steps x 5 d-tiles ----
    v8f cacc[5] = {};
    const h16* vbase = vt + (size_t)(b * HEADS + h) * DV * NPAD;
#pragma unroll
    for (int kk = 0; kk < 7; ++kk) {
        const float* sr = &S[laneLo][kk * 32];
        v16h a;
#pragma unroll
        for (int e = 0; e < 8; ++e) {
            a[e]     = (h16)sr[8 * laneHi + e];
            a[e + 8] = (h16)sr[16 + 8 * laneHi + e];
        }
#pragma unroll
        for (int dt = 0; dt < 5; ++dt) {
            const h16* vrow = vbase + (size_t)(dt * 16 + laneLo) * NPAD + kk * 32;
            v16h bf = load_frag_b(vrow, laneHi, true);
            cacc[dt] = WMMA_F16(a, bf, cacc[dt]);
        }
    }
#pragma unroll
    for (int dt = 0; dt < 5; ++dt) {
        int col = h * DV + dt * 16 + laneLo;
#pragma unroll
        for (int r = 0; r < 8; ++r) {
            int rn = ntile * 16 + 8 * laneHi + r;
            if (rn < SEQ)
                ctxh[(size_t)(b * SEQ + rn) * (HEADS * DV) + col] = (h16)cacc[dt][r];
        }
    }
}

// ---------------- GEMM2: out = ctx @ W_proj + b (f32 out) ----------------
// 32x64 per wave, 2-phase double-buffered K loop (K=640, 20 steps of 32).
// waves = 394*12 = 4728 -> 591 blocks of 256.
__global__ __launch_bounds__(256) void gemm_proj_kernel(
    const h16* __restrict__ ctxh, const h16* __restrict__ wt,
    const float* __restrict__ bias, float* __restrict__ out) {
    int wave = (int)((blockIdx.x * blockDim.x + threadIdx.x) >> 5);
    int lane = threadIdx.x & 31;
    int laneLo = lane & 15, laneHi = lane >> 4;
    int mpair = wave / 12;           // 0..393
    int ngrp  = wave % 12;           // 64 cols each

    const int KDIM = HEADS * DV;     // 640
    const h16* arow0 = ctxh + (size_t)(mpair * 32 + laneLo) * KDIM;
    const h16* arow1 = arow0 + (size_t)16 * KDIM;
    const h16* bcol  = wt + (size_t)(ngrp * 64 + laneLo) * KDIM;

    FragSet s0, s1;
    load_set(s0, arow0, arow1, bcol, KDIM, 0, laneHi);
    load_set(s1, arow0, arow1, bcol, KDIM, 32, laneHi);

    v8f acc[8] = {};
#pragma unroll 1
    for (int kb = 64; kb < KDIM; kb += 64) {
        mma_set(acc, s0);
        load_set(s0, arow0, arow1, bcol, KDIM, kb, laneHi);
        mma_set(acc, s1);
        load_set(s1, arow0, arow1, bcol, KDIM, kb + 32, laneHi);
    }
    mma_set(acc, s0);
    mma_set(acc, s1);

#pragma unroll
    for (int mi = 0; mi < 2; ++mi)
#pragma unroll
        for (int j = 0; j < 4; ++j) {
            int col = ngrp * 64 + j * 16 + laneLo;
            float bv = bias[col];
#pragma unroll
            for (int r = 0; r < 8; ++r) {
                int m = mpair * 32 + mi * 16 + 8 * laneHi + r;
                out[(size_t)m * CIN + col] = acc[mi * 4 + j][r] + bv;
            }
        }
}

// ---------------- host launcher ----------------
extern "C" void kernel_launch(void* const* d_in, const int* in_sizes, int n_in,
                              void* d_out, int out_size, void* d_ws, size_t ws_size,
                              hipStream_t stream) {
    const float* x     = (const float*)d_in[0];
    const float* Wqkv  = (const float*)d_in[1];
    const float* Wproj = (const float*)d_in[2];
    const float* bproj = (const float*)d_in[3];

    float* out  = (float*)d_out;                    // [12608, 768]
    float* attn = out + (size_t)MROWS * CIN;        // [64, 8, 197, 197]

    char* ws = (char*)d_ws;
    h16* xh     = (h16*)(ws + 0);
    h16* wqkvt  = (h16*)(ws + 19365888);
    h16* wprojt = (h16*)(ws + 21921792);
    h16* qkvh   = (h16*)(ws + 22904832);
    h16* vt     = (h16*)(ws + 64864256);
    h16* ctxh   = (h16*)(ws + 83214336);

    cvt_x_kernel    <<<37824, 256, 0, stream>>>(x, xh);
    cvt_wqkv_kernel <<< 4992, 256, 0, stream>>>(Wqkv, wqkvt);
    cvt_wproj_kernel<<< 1920, 256, 0, stream>>>(Wproj, wprojt);

    gemm_qkv_kernel <<< 2561, 128, 0, stream>>>(xh, wqkvt, qkvh);   // 394*26 waves
    build_vt_kernel <<<35840, 256, 0, stream>>>(qkvh, vt);

    attn_fused_kernel<<<dim3(13, HEADS, BATCH), 32, 0, stream>>>(qkvh, vt, attn, ctxh);

    gemm_proj_kernel<<<  591, 256, 0, stream>>>(ctxh, wprojt, bproj, out); // 394*12 waves
}
